// LoftQLinear_65687229825366
// MI455X (gfx1250) — compile-verified, matrix operand order checked by reading, not run on hardware
//
#include <hip/hip_runtime.h>

typedef __attribute__((ext_vector_type(16))) __bf16 bf16x16;
typedef __attribute__((ext_vector_type(8)))  float  v8f;

#define BM 128
#define BN 128
#define BK 64
#define LDK (BK + 8)   // padded LDS row stride (144 B, 16B-aligned) to dodge bank conflicts

// ---------------------------------------------------------------------------
// CDNA5 async global->LDS path (ASYNCcnt), guarded so the file always compiles.
// Builtin signature (from hipcc diagnostic): param0 = v4i* in global/device AS,
// param1 = v4i* in LDS AS, then imm offset, imm cpol.
// ---------------------------------------------------------------------------
#if __has_builtin(__builtin_amdgcn_global_load_async_to_lds_b128) && \
    __has_builtin(__builtin_amdgcn_s_wait_asynccnt)
#define USE_ASYNC_LDS 1
#else
#define USE_ASYNC_LDS 0
#endif

#if USE_ASYNC_LDS
typedef int v4i_t __attribute__((vector_size(16)));
typedef __attribute__((address_space(1))) v4i_t* gv4i_p;
typedef __attribute__((address_space(3))) v4i_t* lv4i_p;
__device__ __forceinline__ void async_cp16(const void* g, void* l) {
  // AS1 pointer has the same bit pattern as the generic global pointer.
  // AS3 pointer is the low 32 bits of the generic (aperture) address.
  gv4i_p gp = (gv4i_p)(unsigned long long)(size_t)g;
  lv4i_p lp = (lv4i_p)(unsigned int)(size_t)l;
  __builtin_amdgcn_global_load_async_to_lds_b128(gp, lp, 0, 0);
}
#endif

// ---------------------------------------------------------------------------
// Prepass 1: f32 -> bf16 conversion of x (keeps x resident in L2 as bf16)
// ---------------------------------------------------------------------------
__global__ __launch_bounds__(256)
void loftq_cvt_f32_bf16(const float* __restrict__ in, __bf16* __restrict__ out, size_t n) {
  size_t i = ((size_t)blockIdx.x * blockDim.x + threadIdx.x) * 4;
  if (i + 4 <= n) {
    const float4 v = *(const float4*)(in + i);
    __bf16 o[4];
    o[0] = (__bf16)v.x; o[1] = (__bf16)v.y; o[2] = (__bf16)v.z; o[3] = (__bf16)v.w;
    *(uint2*)(out + i) = *(const uint2*)o;
  }
}

// ---------------------------------------------------------------------------
// Prepass 2: effW[n,k] = W_q[n,k] + (1/16) * sum_r A[n,r]*B[r,k]   (bf16 out)
// ---------------------------------------------------------------------------
__global__ __launch_bounds__(256)
void loftq_effw_bf16(const float* __restrict__ Wq, const float* __restrict__ A,
                     const float* __restrict__ Bm, __bf16* __restrict__ wb,
                     int K, int R) {
  const int n = blockIdx.x;
  __shared__ float a[16];
  if ((int)threadIdx.x < R) a[threadIdx.x] = A[(size_t)n * R + threadIdx.x] * (1.0f / 16.0f);
  __syncthreads();
  for (int k = threadIdx.x; k < K; k += blockDim.x) {
    float acc = Wq[(size_t)n * K + k];
#pragma unroll
    for (int r = 0; r < 16; ++r) acc += a[r] * Bm[(size_t)r * K + k];
    wb[(size_t)n * K + k] = (__bf16)acc;
  }
}

// ---------------------------------------------------------------------------
// Fragment load (16x32 bf16 operand, CDNA5 16-bit A-matrix layout):
// lane L in [0,15]  -> row M=L,    K = 0..7  and 16..23
// lane L in [16,31] -> row M=L-16, K = 8..15 and 24..31
// ---------------------------------------------------------------------------
__device__ __forceinline__ bf16x16 load_frag(const __bf16* base, int lane) {
  const int r  = lane & 15;
  const int ko = (lane >> 4) << 3;            // 0 or 8
  const __bf16* p = base + r * LDK + ko;
  union { uint4 u[2]; bf16x16 v; } t;
  t.u[0] = *(const uint4*)(p);
  t.u[1] = *(const uint4*)(p + 16);
  return t.v;
}

// ---------------------------------------------------------------------------
// Main GEMM: out[m,n] = sum_k X[m,k]*W[n,k] + bias[n]
// 256 threads = 8 waves; block tile 128x128x64; wave = 32(M) x 64(N).
// 16 x v_wmma_f32_16x16x32_bf16 per k-tile, double-buffered LDS.
// ---------------------------------------------------------------------------
__global__ __launch_bounds__(256)
void loftq_gemm_bf16_wmma(const __bf16* __restrict__ X, const __bf16* __restrict__ W,
                          const float* __restrict__ bias, float* __restrict__ out,
                          int M, int K, int N) {
  __shared__ __bf16 sx[2][BM][LDK];
  __shared__ __bf16 sw[2][BN][LDK];

  const int tid  = threadIdx.x;
  const int lane = tid & 31;
  const int wave = tid >> 5;
  const int wm = (wave >> 1) * 32;   // 0,32,64,96
  const int wn = (wave & 1) * 64;    // 0,64

  const int m0 = blockIdx.y * BM;
  const int n0 = blockIdx.x * BN;

  // Staging coords: each thread moves 32 bf16 (4 x b128) per tile per matrix.
  const int lr = tid >> 1;            // 0..127
  const int lc = (tid & 1) << 5;      // 0 or 32

  const __bf16* xg = X + (size_t)(m0 + lr) * K + lc;
  const __bf16* wg = W + (size_t)(n0 + lr) * K + lc;

  const v8f vzero = {0.f, 0.f, 0.f, 0.f, 0.f, 0.f, 0.f, 0.f};
  v8f acc[2][4];
#pragma unroll
  for (int i = 0; i < 2; ++i)
#pragma unroll
    for (int j = 0; j < 4; ++j) acc[i][j] = vzero;

  const int NK = K / BK;

#define COMPUTE_TILE(BUFIDX)                                                      \
  do {                                                                            \
    _Pragma("unroll")                                                             \
    for (int s = 0; s < 2; ++s) {                                                 \
      bf16x16 af[2], bfr[4];                                                      \
      _Pragma("unroll")                                                           \
      for (int i = 0; i < 2; ++i)                                                 \
        af[i] = load_frag(&sx[BUFIDX][wm + i * 16][s * 32], lane);                \
      _Pragma("unroll")                                                           \
      for (int j = 0; j < 4; ++j)                                                 \
        bfr[j] = load_frag(&sw[BUFIDX][wn + j * 16][s * 32], lane);               \
      _Pragma("unroll")                                                           \
      for (int i = 0; i < 2; ++i)                                                 \
        _Pragma("unroll")                                                         \
        for (int j = 0; j < 4; ++j)                                               \
          acc[i][j] = __builtin_amdgcn_wmma_f32_16x16x32_bf16(                    \
              false, af[i], false, bfr[j], (short)0, acc[i][j], false, false);    \
    }                                                                             \
  } while (0)

#if USE_ASYNC_LDS
  // ---------------- async global->LDS pipeline (ASYNCcnt) ----------------
  {
    __bf16* lx = &sx[0][lr][lc];
    __bf16* lw = &sw[0][lr][lc];
#pragma unroll
    for (int q = 0; q < 4; ++q) {
      async_cp16(xg + q * 8, lx + q * 8);
      async_cp16(wg + q * 8, lw + q * 8);
    }
  }
  for (int kt = 0; kt < NK; ++kt) {
    const int  buf  = kt & 1;
    const bool more = (kt + 1) < NK;
    if (more) {
      const __bf16* xn = xg + (size_t)(kt + 1) * BK;
      const __bf16* wn2 = wg + (size_t)(kt + 1) * BK;
      __bf16* lx = &sx[buf ^ 1][lr][lc];
      __bf16* lw = &sw[buf ^ 1][lr][lc];
#pragma unroll
      for (int q = 0; q < 4; ++q) {
        async_cp16(xn + q * 8, lx + q * 8);
        async_cp16(wn2 + q * 8, lw + q * 8);
      }
      __builtin_amdgcn_s_wait_asynccnt(8);  // tile kt landed; kt+1 still in flight
    } else {
      __builtin_amdgcn_s_wait_asynccnt(0);
    }
    __syncthreads();          // everyone's async portion of tile kt visible
    COMPUTE_TILE(buf);
    __syncthreads();          // all waves done reading buf before it is re-filled
  }
#else
  // ---------------- register-staged fallback ----------------
  {
    uint4 r[8];
#pragma unroll
    for (int q = 0; q < 4; ++q) {
      r[q]     = *(const uint4*)(xg + q * 8);
      r[4 + q] = *(const uint4*)(wg + q * 8);
    }
#pragma unroll
    for (int q = 0; q < 4; ++q) {
      *(uint4*)&sx[0][lr][lc + 8 * q] = r[q];
      *(uint4*)&sw[0][lr][lc + 8 * q] = r[4 + q];
    }
  }
  __syncthreads();

  for (int kt = 0; kt < NK; ++kt) {
    const int  buf  = kt & 1;
    const bool more = (kt + 1) < NK;

    uint4 r[8];
    if (more) {
      const __bf16* xn  = xg + (size_t)(kt + 1) * BK;
      const __bf16* wn2 = wg + (size_t)(kt + 1) * BK;
#pragma unroll
      for (int q = 0; q < 4; ++q) {
        r[q]     = *(const uint4*)(xn + q * 8);
        r[4 + q] = *(const uint4*)(wn2 + q * 8);
      }
      if (kt + 2 < NK) {
        __builtin_prefetch(xg + (size_t)(kt + 2) * BK, 0, 0);  // global_prefetch_b8
        __builtin_prefetch(wg + (size_t)(kt + 2) * BK, 0, 0);
      }
    }

    COMPUTE_TILE(buf);

    if (more) {
      __syncthreads();
      const int nb = buf ^ 1;
#pragma unroll
      for (int q = 0; q < 4; ++q) {
        *(uint4*)&sx[nb][lr][lc + 8 * q] = r[q];
        *(uint4*)&sw[nb][lr][lc + 8 * q] = r[4 + q];
      }
      __syncthreads();
    }
  }
#endif

  // Epilogue: C/D layout -> lanes 0-15: M=e, lanes 16-31: M=e+8, N = lane&15.
  const int cn = lane & 15;
  const int mh = (lane >> 4) << 3;
#pragma unroll
  for (int i = 0; i < 2; ++i) {
    const int gm0 = m0 + wm + i * 16 + mh;
#pragma unroll
    for (int j = 0; j < 4; ++j) {
      const int gn = n0 + wn + j * 16 + cn;
      const float bv = bias[gn];
      float* p = out + (size_t)gm0 * N + gn;
#pragma unroll
      for (int e = 0; e < 8; ++e) { *p = acc[i][j][e] + bv; p += N; }
    }
  }
}

// ---------------------------------------------------------------------------
// Host launcher
// ---------------------------------------------------------------------------
extern "C" void kernel_launch(void* const* d_in, const int* in_sizes, int n_in,
                              void* d_out, int out_size, void* d_ws, size_t ws_size,
                              hipStream_t stream) {
  const float* x    = (const float*)d_in[0];
  const float* Wq   = (const float*)d_in[1];
  const float* A    = (const float*)d_in[2];
  const float* Bm   = (const float*)d_in[3];
  const float* bias = (const float*)d_in[4];
  float* out = (float*)d_out;

  const int R    = 16;
  const int Dout = in_sizes[4];                        // bias length
  const int Din  = in_sizes[3] / R;                    // B is (R, Din)
  const size_t M = (size_t)in_sizes[0] / (size_t)Din;  // B_*S

  __bf16* xb = (__bf16*)d_ws;                // M x Din bf16
  __bf16* wb = xb + (size_t)M * Din;         // Dout x Din bf16

  const size_t nx = M * (size_t)Din;
  {
    const size_t threads = nx / 4;
    const int blocks = (int)((threads + 255) / 256);
    loftq_cvt_f32_bf16<<<blocks, 256, 0, stream>>>(x, xb, nx);
  }

  loftq_effw_bf16<<<Dout, 256, 0, stream>>>(Wq, A, Bm, wb, Din, R);

  dim3 grid(Dout / BN, (unsigned)(M / BM));
  loftq_gemm_bf16_wmma<<<grid, 256, 0, stream>>>(xb, wb, bias, out,
                                                 (int)M, Din, Dout);
}
